// KDEHyperGraphCompute_38027640438849
// MI455X (gfx1250) — matrix operand order repaired and checked by reading, not compiled, not run on hardware
//
#include <hip/hip_runtime.h>
#include <hip/hip_bf16.h>

typedef __attribute__((ext_vector_type(16))) __bf16 v16bf;
typedef __attribute__((ext_vector_type(8)))  float  v8f;

#define B_DIM 16
#define C_DIM 256
#define T_DIM 6400
#define M_DIM 48
#define TILE_T 128
#define NBLK (T_DIM / TILE_T)   /* 50 */
#define TAU_INV 10.0f

static __device__ __forceinline__ v8f zero8() {
  v8f z;
#pragma unroll
  for (int i = 0; i < 8; ++i) z[i] = 0.f;
  return z;
}

static __device__ __forceinline__ float block_reduce_sum(float v, float* red) {
  int tid = threadIdx.x;
  red[tid] = v;
  __syncthreads();
  for (int s = 128; s > 0; s >>= 1) {
    if (tid < s) red[tid] += red[tid + s];
    __syncthreads();
  }
  float r = red[0];
  __syncthreads();
  return r;
}

// ---------------------------------------------------------------------------
// BN folding: scale = gamma/sqrt(var+eps); bias = beta - mean*scale
// ---------------------------------------------------------------------------
__global__ __launch_bounds__(256)
void bnprep_kernel(const float* __restrict__ g, const float* __restrict__ be,
                   const float* __restrict__ mn, const float* __restrict__ vr,
                   float* __restrict__ scale, float* __restrict__ bias) {
  int o = threadIdx.x;
  float sc = g[o] * rsqrtf(vr[o] + 1e-5f);
  scale[o] = sc;
  bias[o] = be[o] - mn[o] * sc;
}

// ---------------------------------------------------------------------------
// Initial centers: normalized token rows at linspace indices (bf16 store)
// ---------------------------------------------------------------------------
__global__ __launch_bounds__(256)
void init_centers_kernel(const float* __restrict__ x, __bf16* __restrict__ cen) {
  __shared__ float red[256];
  int m = blockIdx.x, b = blockIdx.y, c = threadIdx.x;
  int tidx = (int)((double)m * (double)(T_DIM - 1) / 47.0);
  float val = x[((size_t)b * C_DIM + c) * T_DIM + tidx];
  float nrm = block_reduce_sum(val * val, red);
  float inv = 1.f / fmaxf(sqrtf(nrm), 1e-6f);
  cen[((size_t)b * M_DIM + m) * C_DIM + c] = (__bf16)(val * inv);
}

// ---------------------------------------------------------------------------
// One soft-k-means iteration, fused:
//   sim = tok . cen^T   (WMMA bf16, K=channels, per-wave 16-token tile)
//   softmax over 48 centers (shfl_xor within 16-lane halves)
//   num = w^T . tok     (WMMA bf16, K=tokens)   -> split-K partials
// ---------------------------------------------------------------------------
__global__ __launch_bounds__(256)
void assign_kernel(const float* __restrict__ x, const __bf16* __restrict__ cen,
                   float* __restrict__ w_buf, float* __restrict__ num_part,
                   float* __restrict__ den_part, int store_w) {
  __shared__ float w_lds[TILE_T * M_DIM];   // 24 KB
  const int b = blockIdx.y;
  const int blk = blockIdx.x;
  const int t0 = blk * TILE_T;
  const int tid = threadIdx.x;
  const int wv = tid >> 5;
  const int lane = tid & 31;
  const int hi = lane >> 4;
  const int r = lane & 15;
  const int t0w = t0 + wv * 16;

  // ---------------- sim phase ----------------
  v8f acc[3];
  acc[0] = zero8(); acc[1] = zero8(); acc[2] = zero8();
  float sq = 0.f;
  const float* xbase = x + (size_t)b * C_DIM * T_DIM + (t0w + r);
  const __bf16* cb_base = cen + (size_t)b * M_DIM * C_DIM;

#pragma unroll
  for (int cb = 0; cb < 8; ++cb) {
    // A = token tile [16 x 32] bf16 (row = token r, K = channels)
    v16bf a;
    const float* xa = xbase + (size_t)(32 * cb + 8 * hi) * T_DIM;
#pragma unroll
    for (int v = 0; v < 4; ++v)
#pragma unroll
      for (int j = 0; j < 2; ++j) {
        float val = xa[(size_t)(2 * v + j) * T_DIM];
        a[2 * v + j] = (__bf16)val;
        sq += val * val;
      }
#pragma unroll
    for (int v = 4; v < 8; ++v)
#pragma unroll
      for (int j = 0; j < 2; ++j) {
        float val = xa[(size_t)(16 + 2 * (v - 4) + j) * T_DIM];
        a[2 * v + j] = (__bf16)val;
        sq += val * val;
      }
    // B = centers [32 x 16] bf16, two 16B runs per lane (k=8hi..+7, 16+8hi..+7)
#pragma unroll
    for (int mt = 0; mt < 3; ++mt) {
      union { v16bf v; uint4 u[2]; } bp;
      const uint4* q = reinterpret_cast<const uint4*>(
          cb_base + (size_t)(16 * mt + r) * C_DIM + 32 * cb + 8 * hi);
      bp.u[0] = q[0]; bp.u[1] = q[2];
      acc[mt] = __builtin_amdgcn_wmma_f32_16x16x32_bf16(
          false, a, false, bp.v, (short)0, acc[mt], false, false);
    }
  }

  // per-token squared norm: halves hold disjoint channel sets
  float sqtot = sq + __shfl_xor(sq, 16);

  // ---------------- softmax over 48 centers ----------------
#pragma unroll
  for (int v = 0; v < 8; ++v) {
    float sn = __shfl(sqtot, v + 8 * hi);        // norm of token row (v+8hi)
    float inv = 1.f / fmaxf(sqrtf(sn), 1e-6f);
    float s0 = acc[0][v] * inv;
    float s1 = acc[1][v] * inv;
    float s2 = acc[2][v] * inv;
    float mx = fmaxf(fmaxf(s0, s1), s2);
#pragma unroll
    for (int off = 1; off < 16; off <<= 1) mx = fmaxf(mx, __shfl_xor(mx, off));
    float e0 = __expf((s0 - mx) * TAU_INV);
    float e1 = __expf((s1 - mx) * TAU_INV);
    float e2 = __expf((s2 - mx) * TAU_INV);
    float es = e0 + e1 + e2;
#pragma unroll
    for (int off = 1; off < 16; off <<= 1) es += __shfl_xor(es, off);
    float einv = 1.f / es;
    int row = (wv * 16 + v + 8 * hi) * M_DIM + r;
    w_lds[row]      = e0 * einv;
    w_lds[row + 16] = e1 * einv;
    w_lds[row + 32] = e2 * einv;
  }
  __syncthreads();

  // den partial (per-block token sum of w)
  if (tid < M_DIM) {
    float d = 0.f;
    for (int t = 0; t < TILE_T; ++t) d += w_lds[t * M_DIM + tid];
    den_part[((size_t)blk * B_DIM + b) * M_DIM + tid] = d;
  }
  if (store_w) {
    float* wb = w_buf + ((size_t)b * T_DIM + t0) * M_DIM;
    for (int i = tid; i < TILE_T * M_DIM; i += 256) wb[i] = w_lds[i];
  }

  // ---------------- num = w^T . tok (split-K partial) ----------------
  // waves split the 16 channel tiles; K = 128 tokens in 4 chunks of 32
#pragma unroll
  for (int cti = 0; cti < 2; ++cti) {
    int ct = wv + 8 * cti;
    int c = ct * 16 + r;
    const float* xp0 = x + ((size_t)b * C_DIM + c) * T_DIM + t0 + 16 * hi;
#pragma unroll
    for (int mt = 0; mt < 3; ++mt) {
      v8f acn = zero8();
#pragma unroll
      for (int kc = 0; kc < 4; ++kc) {
        v16bf a;  // A[m=center row][k=token]
#pragma unroll
        for (int v = 0; v < 4; ++v)
#pragma unroll
          for (int j = 0; j < 2; ++j)
            a[2 * v + j] = (__bf16)w_lds[(32 * kc + 16 * hi + 2 * v + j) * M_DIM + 16 * mt + r];
#pragma unroll
        for (int v = 4; v < 8; ++v)
#pragma unroll
          for (int j = 0; j < 2; ++j)
            a[2 * v + j] = (__bf16)w_lds[(32 * kc + 16 * hi + 8 + 2 * (v - 4) + j) * M_DIM + 16 * mt + r];
        v16bf bb; // B[k=token][n=channel] : 16 consecutive f32 = one 64B line
        const float* xp = xp0 + 32 * kc;
#pragma unroll
        for (int e = 0; e < 16; ++e) bb[e] = (__bf16)xp[e];
        acn = __builtin_amdgcn_wmma_f32_16x16x32_bf16(
            false, a, false, bb, (short)0, acn, false, false);
      }
      float* np = num_part + (((size_t)blk * B_DIM + b) * M_DIM + 16 * mt + 8 * hi) * C_DIM + c;
#pragma unroll
      for (int v = 0; v < 8; ++v) np[(size_t)v * C_DIM] = acn[v];
    }
  }
}

// ---------------------------------------------------------------------------
// Reduce split-K partials -> normalized centers (l2norm(num/den)=l2norm(num))
// ---------------------------------------------------------------------------
__global__ __launch_bounds__(256)
void reduce_centers_kernel(const float* __restrict__ num_part, __bf16* __restrict__ cen) {
  __shared__ float red[256];
  int m = blockIdx.x, b = blockIdx.y, c = threadIdx.x;
  float s = 0.f;
  for (int blk = 0; blk < NBLK; ++blk)
    s += num_part[(((size_t)blk * B_DIM + b) * M_DIM + m) * C_DIM + c];
  float nrm = block_reduce_sum(s * s, red);
  float inv = 1.f / fmaxf(sqrtf(nrm), 1e-6f);
  cen[((size_t)b * M_DIM + m) * C_DIM + c] = (__bf16)(s * inv);
}

// ---------------------------------------------------------------------------
// Reduce partials -> nodes = num / max(den, 1e-6)
// ---------------------------------------------------------------------------
__global__ __launch_bounds__(256)
void finalize_nodes_kernel(const float* __restrict__ num_part,
                           const float* __restrict__ den_part,
                           float* __restrict__ nodes) {
  __shared__ float dsh;
  int m = blockIdx.x, b = blockIdx.y, c = threadIdx.x;
  if (c == 0) {
    float d = 0.f;
    for (int blk = 0; blk < NBLK; ++blk)
      d += den_part[((size_t)blk * B_DIM + b) * M_DIM + m];
    dsh = fmaxf(d, 1e-6f);
  }
  __syncthreads();
  float s = 0.f;
  for (int blk = 0; blk < NBLK; ++blk)
    s += num_part[(((size_t)blk * B_DIM + b) * M_DIM + m) * C_DIM + c];
  nodes[((size_t)b * M_DIM + m) * C_DIM + c] = s / dsh;
}

// ---------------------------------------------------------------------------
// KDE hypergraph on 48 nodes: y = A(nodes) @ nodes   (one block per batch)
// ---------------------------------------------------------------------------
__global__ __launch_bounds__(256)
void hyper_kernel(const float* __restrict__ nodes, float* __restrict__ y) {
  __shared__ float nd[M_DIM * C_DIM];   // 48 KB
  __shared__ float buf[M_DIM * M_DIM];  // d2, then A (9 KB)
  __shared__ float red[256];
  __shared__ float sq[M_DIM], rho[M_DIM], De[M_DIM], Dv[M_DIM];
  __shared__ float inv2h2;
  int b = blockIdx.x, tid = threadIdx.x;
  const float* nb = nodes + (size_t)b * M_DIM * C_DIM;
  for (int i = tid; i < M_DIM * C_DIM; i += 256) nd[i] = nb[i];
  __syncthreads();
  if (tid < M_DIM) {
    float s = 0.f;
    for (int c = 0; c < C_DIM; ++c) { float v = nd[tid * C_DIM + c]; s += v * v; }
    sq[tid] = s;
  }
  __syncthreads();
  float dsum = 0.f;
  for (int p = tid; p < M_DIM * M_DIM; p += 256) {
    int i = p / M_DIM, j = p % M_DIM;
    float dot = 0.f;
    for (int c = 0; c < C_DIM; ++c) dot += nd[i * C_DIM + c] * nd[j * C_DIM + c];
    float d2 = fmaxf(sq[i] + sq[j] - 2.f * dot, 0.f);
    buf[p] = d2;
    if (d2 > 0.f) dsum += sqrtf(d2);
  }
  float tot = block_reduce_sum(dsum, red);
  if (tid == 0) {
    float h = fmaxf(tot / (float)(M_DIM * (M_DIM - 1)), 1e-6f);
    inv2h2 = 1.f / (2.f * h * h);
  }
  __syncthreads();
  if (tid < M_DIM) {
    float s = 0.f;
    for (int j = 0; j < M_DIM; ++j) s += __expf(-buf[tid * M_DIM + j] * inv2h2);
    rho[tid] = s;
  }
  __syncthreads();
  if (tid < M_DIM) {
    float de = 1.f, dv = 1.f;
    float rj = rho[tid];
    for (int i = 0; i < M_DIM; ++i) {
      if (rho[i] > rj) de += 1.f;
      if (rj > rho[i]) dv += 1.f;
    }
    De[tid] = de; Dv[tid] = dv;
  }
  __syncthreads();
  for (int p = tid; p < M_DIM * M_DIM; p += 256) {
    int i = p / M_DIM, k = p % M_DIM;
    float ri = rho[i], rk = rho[k];
    float s = 0.f;
    for (int j = 0; j < M_DIM; ++j) {
      float rj = rho[j];
      bool hij = (ri > rj) || (i == j);
      bool hkj = (rk > rj) || (k == j);
      if (hij && hkj) s += 1.f / De[j];
    }
    buf[p] = s * rsqrtf(Dv[i] * Dv[k]);
  }
  __syncthreads();
  float* yb = y + (size_t)b * M_DIM * C_DIM;
  for (int p = tid; p < M_DIM * C_DIM; p += 256) {
    int i = p / C_DIM, c = p % C_DIM;
    float s = 0.f;
    for (int k = 0; k < M_DIM; ++k) s += buf[i * M_DIM + k] * nd[k * C_DIM + c];
    yb[p] = s;
  }
}

// ---------------------------------------------------------------------------
// nodes_out = silu(y @ proj_w^T)
// ---------------------------------------------------------------------------
__global__ __launch_bounds__(256)
void proj_kernel(const float* __restrict__ y, const float* __restrict__ pw,
                 float* __restrict__ nout) {
  __shared__ float row[C_DIM];
  int m = blockIdx.x, b = blockIdx.y, o = threadIdx.x;
  row[o] = y[((size_t)b * M_DIM + m) * C_DIM + o];
  __syncthreads();
  const float* w = pw + (size_t)o * C_DIM;
  float s = 0.f;
  for (int c = 0; c < C_DIM; ++c) s += row[c] * w[c];
  s = s / (1.f + __expf(-s));
  nout[((size_t)b * M_DIM + m) * C_DIM + o] = s;
}

// ---------------------------------------------------------------------------
// Fold 1x1 conv into nodes: ncT[b][o][m] = sum_c nodes_out[b][m][c]*conv_w[o][c]
// transposed + bf16 + m padded to 64 so phase C's B-matrix is two 16B runs/lane.
// ---------------------------------------------------------------------------
__global__ __launch_bounds__(256)
void conv_kernel(const float* __restrict__ nout, const float* __restrict__ cw,
                 __bf16* __restrict__ ncT) {
  __shared__ float row[C_DIM];
  int m = blockIdx.x, b = blockIdx.y, o = threadIdx.x;
  if (m >= M_DIM) {
    ncT[((size_t)b * C_DIM + o) * 64 + m] = (__bf16)0.f;
    return;
  }
  row[o] = nout[((size_t)b * M_DIM + m) * C_DIM + o];
  __syncthreads();
  const float* w = cw + (size_t)o * C_DIM;
  float s = 0.f;
  for (int c = 0; c < C_DIM; ++c) s += row[c] * w[c];
  ncT[((size_t)b * C_DIM + o) * 64 + m] = (__bf16)s;
}

// ---------------------------------------------------------------------------
// Phase C: z = w @ ncT^T (WMMA, K=48 padded to 64), BN + SiLU + residual.
// ---------------------------------------------------------------------------
__global__ __launch_bounds__(256)
void final_kernel(const float* __restrict__ x, const float* __restrict__ w_buf,
                  const __bf16* __restrict__ ncT, const float* __restrict__ scale,
                  const float* __restrict__ bias, float* __restrict__ out) {
  __shared__ float w_lds[TILE_T * M_DIM];   // 24 KB
  const int b = blockIdx.y, blk = blockIdx.x, tid = threadIdx.x;
  const int t0 = blk * TILE_T;
  const int wv = tid >> 5, lane = tid & 31, hi = lane >> 4, r = lane & 15;
  const float* wb = w_buf + ((size_t)b * T_DIM + t0) * M_DIM;
  for (int i = tid; i < TILE_T * M_DIM; i += 256) w_lds[i] = wb[i];
  __syncthreads();
  const int trow = wv * 16;
#pragma unroll 2
  for (int nt = 0; nt < 16; ++nt) {
    v8f acc = zero8();
#pragma unroll
    for (int kc = 0; kc < 2; ++kc) {
      v16bf a;  // A[row=token r][k=center], zero-padded past 48
#pragma unroll
      for (int v = 0; v < 8; ++v)
#pragma unroll
        for (int j = 0; j < 2; ++j) {
          int k = 32 * kc + ((v < 4) ? (8 * hi + 2 * v + j)
                                     : (16 + 8 * hi + 2 * (v - 4) + j));
          float val = (k < M_DIM) ? w_lds[(trow + r) * M_DIM + k] : 0.f;
          a[2 * v + j] = (__bf16)val;
        }
      union { v16bf v; uint4 u[2]; } bp;
      const uint4* q = reinterpret_cast<const uint4*>(
          ncT + ((size_t)b * C_DIM + nt * 16 + r) * 64 + 32 * kc + 8 * hi);
      bp.u[0] = q[0]; bp.u[1] = q[2];
      acc = __builtin_amdgcn_wmma_f32_16x16x32_bf16(
          false, a, false, bp.v, (short)0, acc, false, false);
    }
    int o = nt * 16 + r;
    float sc = scale[o], bi = bias[o];
    const size_t base = ((size_t)b * C_DIM + o) * T_DIM + t0 + trow + 8 * hi;
    const float* xr = x + base;
    float* orow = out + base;
#pragma unroll
    for (int v = 0; v < 8; ++v) {
      float z = acc[v] * sc + bi;
      z = z / (1.f + __expf(-z));
      orow[v] = xr[v] + z;
    }
  }
}

// ---------------------------------------------------------------------------
extern "C" void kernel_launch(void* const* d_in, const int* in_sizes, int n_in,
                              void* d_out, int out_size, void* d_ws, size_t ws_size,
                              hipStream_t stream) {
  const float* x      = (const float*)d_in[0];
  const float* proj_w = (const float*)d_in[1];
  const float* conv_w = (const float*)d_in[2];
  const float* bn_g   = (const float*)d_in[3];
  const float* bn_b   = (const float*)d_in[4];
  const float* bn_m   = (const float*)d_in[5];
  const float* bn_v   = (const float*)d_in[6];
  float* out = (float*)d_out;

  char* ws = (char*)d_ws;
  // workspace layout (bytes, all 256-aligned), total ~62.4 MB
  __bf16* cen      = (__bf16*)(ws + 0);            // 16*48*256*2      = 393216
  float*  w_buf    = (float*)(ws + 393216);        // 16*6400*48*4     = 19660800
  float*  num_part = (float*)(ws + 20054016);      // 50*16*48*256*4   = 39321600
  float*  den_part = (float*)(ws + 59375616);      // 50*16*48*4       = 153600
  float*  nodes    = (float*)(ws + 59529216);      // 16*48*256*4      = 786432
  float*  ybuf     = (float*)(ws + 60315648);      // 786432
  float*  nout     = (float*)(ws + 61102080);      // 786432
  __bf16* ncT      = (__bf16*)(ws + 61888512);     // 16*256*64*2      = 524288
  float*  scale    = (float*)(ws + 62412800);      // 1024
  float*  bias     = (float*)(ws + 62413824);      // 1024

  bnprep_kernel<<<1, 256, 0, stream>>>(bn_g, bn_b, bn_m, bn_v, scale, bias);
  init_centers_kernel<<<dim3(M_DIM, B_DIM), 256, 0, stream>>>(x, cen);
  for (int it = 0; it < 3; ++it) {
    assign_kernel<<<dim3(NBLK, B_DIM), 256, 0, stream>>>(
        x, cen, w_buf, num_part, den_part, (it == 2) ? 1 : 0);
    if (it < 2)
      reduce_centers_kernel<<<dim3(M_DIM, B_DIM), 256, 0, stream>>>(num_part, cen);
  }
  finalize_nodes_kernel<<<dim3(M_DIM, B_DIM), 256, 0, stream>>>(num_part, den_part, nodes);
  hyper_kernel<<<B_DIM, 256, 0, stream>>>(nodes, ybuf);
  proj_kernel<<<dim3(M_DIM, B_DIM), 256, 0, stream>>>(ybuf, proj_w, nout);
  conv_kernel<<<dim3(64, B_DIM), 256, 0, stream>>>(nout, conv_w, ncT);
  final_kernel<<<dim3(NBLK, B_DIM), 256, 0, stream>>>(x, w_buf, ncT, scale, bias, out);
}
